// HeterogeneousGATLayer_18468359372805
// MI455X (gfx1250) — compile-verified, matrix operand order checked by reading, not run on hardware
//
#include <hip/hip_runtime.h>
#include <math.h>

// ---------------------------------------------------------------------------
// HeterogeneousGATLayer for MI455X (gfx1250, wave32, WMMA).
//   h   = x @ lin_w.T                      (WMMA bf16 16x16x32, f32 acc)
//   a_s, a_d = per-head dots with att vecs
//   edge softmax: atomic segment-max (monotone uint keys) + atomic denom
//   agg = scatter-add(alpha * h[src])      (global_atomic_add_f32)
//   out = LN((agg + gat_bias) @ proj_w.T + proj_b)   (WMMA + shfl_xor LN)
// ---------------------------------------------------------------------------

typedef __attribute__((ext_vector_type(16))) __bf16 v16bf;
typedef __attribute__((ext_vector_type(8)))  float  v8f;

#define CH        128
#define HEADS     8
#define HEAD_DIM  16
#define NEG_SLOPE 0.2f
#define LN_EPS    1e-5f

// order-preserving float -> uint encoding for atomicMax-based segment max
__device__ __forceinline__ unsigned enc_f32(float f) {
    unsigned b = __float_as_uint(f);
    return (b & 0x80000000u) ? ~b : (b | 0x80000000u);
}
__device__ __forceinline__ float dec_f32(unsigned k) {
    unsigned b = (k & 0x80000000u) ? (k & 0x7FFFFFFFu) : ~k;
    return __uint_as_float(b);
}

__device__ __forceinline__ void edge_pair(const int* __restrict__ ei, int E,
                                          int e, int& s, int& d) {
    if (e < E) { s = ei[e]; d = ei[E + e]; }
    else       { s = e - E; d = s; }          // self loops appended
}

__device__ __forceinline__ float leaky(float x) {
    return x > 0.f ? x : NEG_SLOPE * x;
}

// ---------------------------------------------------------------------------
// Out[N,128] = (A[N,128] (+bias_in)) @ W.T (+bias_out),  W is [128 out,128 in]
// 8 waves/block, each wave owns a 16-row tile and all 8 column tiles.
// Weights staged once per block in LDS as bf16 (32 KB of 320 KB WGP LDS).
// Fragment layouts per CDNA5 ISA 7.12.2 (16-bit A/B: lane = row/col & 15,
// elements [0..7] = K run at half*8, [8..15] = K run at 16+half*8).
// HAS_BIAS is compile-time: no per-element branches in the hot loop.
// ---------------------------------------------------------------------------
template<bool HAS_BIAS>
__global__ __launch_bounds__(256)
void gat_gemm128_wmma(const float* __restrict__ A, const float* __restrict__ W,
                      const float* __restrict__ bias_in,
                      const float* __restrict__ bias_out,
                      float* __restrict__ Out, int N) {
    __shared__ __bf16 wlds[CH * CH];
    int tid = threadIdx.x;
    for (int i = tid; i < CH * CH; i += 256) wlds[i] = (__bf16)W[i];
    __syncthreads();

    int wave = tid >> 5;
    int lane = tid & 31;
    int half = lane >> 4;      // 0: lanes 0-15, 1: lanes 16-31
    int l15  = lane & 15;

    int tiles = (N + 15) >> 4;
    int tile  = blockIdx.x * 8 + wave;
    if (tile >= tiles) return;
    int row0 = tile << 4;

    int arow = row0 + l15;
    if (arow >= N) arow = N - 1;                 // clamp; stores are guarded
    const float* xr = A + (size_t)arow * CH;

    v8f acc[8] = {};

    #pragma unroll
    for (int kc = 0; kc < 4; ++kc) {             // K = 128 in chunks of 32
        int k0 = kc * 32 + half * 8;             // lane's contiguous K runs
        const float* xp = xr + k0;
        float4 p0 = *(const float4*)(xp);        // K = k0 .. k0+3
        float4 p1 = *(const float4*)(xp + 4);    // K = k0+4 .. k0+7
        float4 p2 = *(const float4*)(xp + 16);   // K = k0+16 .. k0+19
        float4 p3 = *(const float4*)(xp + 20);   // K = k0+20 .. k0+23
        if (HAS_BIAS) {
            const float* bp = bias_in + k0;
            float4 b0 = *(const float4*)(bp);
            float4 b1 = *(const float4*)(bp + 4);
            float4 b2 = *(const float4*)(bp + 16);
            float4 b3 = *(const float4*)(bp + 20);
            p0.x += b0.x; p0.y += b0.y; p0.z += b0.z; p0.w += b0.w;
            p1.x += b1.x; p1.y += b1.y; p1.z += b1.z; p1.w += b1.w;
            p2.x += b2.x; p2.y += b2.y; p2.z += b2.z; p2.w += b2.w;
            p3.x += b3.x; p3.y += b3.y; p3.z += b3.z; p3.w += b3.w;
        }
        v16bf a;
        a[0]  = (__bf16)p0.x; a[1]  = (__bf16)p0.y; a[2]  = (__bf16)p0.z; a[3]  = (__bf16)p0.w;
        a[4]  = (__bf16)p1.x; a[5]  = (__bf16)p1.y; a[6]  = (__bf16)p1.z; a[7]  = (__bf16)p1.w;
        a[8]  = (__bf16)p2.x; a[9]  = (__bf16)p2.y; a[10] = (__bf16)p2.z; a[11] = (__bf16)p2.w;
        a[12] = (__bf16)p3.x; a[13] = (__bf16)p3.y; a[14] = (__bf16)p3.z; a[15] = (__bf16)p3.w;

        #pragma unroll
        for (int t = 0; t < 8; ++t) {            // 8 column tiles of 16
            const __bf16* wr = &wlds[(t * 16 + l15) * CH];  // B col = W row
            v16bf b;
            #pragma unroll
            for (int i = 0; i < 8; ++i) { b[i] = wr[k0 + i]; b[8 + i] = wr[k0 + 16 + i]; }
            acc[t] = __builtin_amdgcn_wmma_f32_16x16x32_bf16(
                false, a, false, b, (short)0, acc[t], false, false);
        }
    }

    #pragma unroll
    for (int t = 0; t < 8; ++t) {
        int col = t * 16 + l15;
        float badd = HAS_BIAS ? bias_out[col] : 0.0f;
        #pragma unroll
        for (int r = 0; r < 8; ++r) {            // C: m = 8*half + r, n = l15
            int row = row0 + half * 8 + r;
            if (row < N) Out[(size_t)row * CH + col] = acc[t][r] + badd;
        }
    }
}

// one thread per (node, head): a_src/a_dst 16-wide dots
__global__ void gat_att_coef(const float* __restrict__ h,
                             const float* __restrict__ att_src,
                             const float* __restrict__ att_dst,
                             float* __restrict__ a_src,
                             float* __restrict__ a_dst, int N) {
    int idx = blockIdx.x * blockDim.x + threadIdx.x;
    if (idx >= N * HEADS) return;
    int n = idx >> 3, hd = idx & 7;
    const float* hp = h + (size_t)n * CH + hd * HEAD_DIM;
    const float* as = att_src + hd * HEAD_DIM;
    const float* ad = att_dst + hd * HEAD_DIM;
    float s = 0.f, d = 0.f;
    #pragma unroll
    for (int i = 0; i < HEAD_DIM; ++i) { float v = hp[i]; s += v * as[i]; d += v * ad[i]; }
    a_src[idx] = s;
    a_dst[idx] = d;
}

// re-run every launch (graph replay): zero agg/denom, -inf max keys
__global__ void gat_init(float* __restrict__ agg, float* __restrict__ denom,
                         unsigned* __restrict__ maxkey, int N) {
    int idx = blockIdx.x * blockDim.x + threadIdx.x;
    if (idx < N * CH) agg[idx] = 0.f;
    if (idx < N * HEADS) { denom[idx] = 0.f; maxkey[idx] = 0u; }
}

__global__ void gat_edge_max(const int* __restrict__ ei,
                             const float* __restrict__ a_src,
                             const float* __restrict__ a_dst,
                             unsigned* __restrict__ maxkey, int E, int N) {
    int gid = blockIdx.x * blockDim.x + threadIdx.x;
    if (gid >= (E + N) * HEADS) return;
    int e = gid >> 3, hd = gid & 7;
    int s, d; edge_pair(ei, E, e, s, d);
    float x = leaky(a_src[s * 8 + hd] + a_dst[d * 8 + hd]);
    atomicMax(&maxkey[d * 8 + hd], enc_f32(x));
}

__global__ void gat_edge_denom(const int* __restrict__ ei,
                               const float* __restrict__ a_src,
                               const float* __restrict__ a_dst,
                               const unsigned* __restrict__ maxkey,
                               float* __restrict__ denom, int E, int N) {
    int gid = blockIdx.x * blockDim.x + threadIdx.x;
    if (gid >= (E + N) * HEADS) return;
    int e = gid >> 3, hd = gid & 7;
    int s, d; edge_pair(ei, E, e, s, d);
    float x = leaky(a_src[s * 8 + hd] + a_dst[d * 8 + hd]);
    float m = dec_f32(maxkey[d * 8 + hd]);
    atomicAdd(&denom[d * 8 + hd], __expf(x - m));
}

// 32 lanes per edge, 4 channels/lane (all same head); scatter via f32 atomics
__global__ __launch_bounds__(256)
void gat_edge_aggregate(const int* __restrict__ ei, const float* __restrict__ h,
                        const float* __restrict__ a_src,
                        const float* __restrict__ a_dst,
                        const unsigned* __restrict__ maxkey,
                        const float* __restrict__ denom,
                        float* __restrict__ agg, int E, int N) {
    int lane = threadIdx.x & 31;
    int e = blockIdx.x * 8 + (threadIdx.x >> 5);
    if (e >= E + N) return;
    int s, d; edge_pair(ei, E, e, s, d);

    const float* hrow = h + (size_t)s * CH;
    __builtin_prefetch(hrow + lane * 4, 0, 1);   // global_prefetch_b8: hide gather

    int hd = lane >> 2;                          // channels 4L..4L+3 share a head
    float x = leaky(a_src[s * 8 + hd] + a_dst[d * 8 + hd]);
    float m = dec_f32(maxkey[d * 8 + hd]);
    float alpha = __expf(x - m) / denom[d * 8 + hd];

    float4 v = ((const float4*)hrow)[lane];
    float* ap = agg + (size_t)d * CH + lane * 4;
    atomicAdd(ap + 0, v.x * alpha);
    atomicAdd(ap + 1, v.y * alpha);
    atomicAdd(ap + 2, v.z * alpha);
    atomicAdd(ap + 3, v.w * alpha);
}

// one wave per row; 4 channels/lane; shfl_xor tree over 32 lanes (wave32)
__global__ __launch_bounds__(256)
void gat_layernorm(float* __restrict__ out, const float* __restrict__ gamma,
                   const float* __restrict__ beta, int N) {
    int lane = threadIdx.x & 31;
    int row  = blockIdx.x * 8 + (threadIdx.x >> 5);
    if (row >= N) return;
    float* rp = out + (size_t)row * CH;
    float v[4], s = 0.f, sq = 0.f;
    #pragma unroll
    for (int i = 0; i < 4; ++i) { v[i] = rp[lane + 32 * i]; s += v[i]; sq += v[i] * v[i]; }
    #pragma unroll
    for (int off = 16; off > 0; off >>= 1) {
        s  += __shfl_xor(s,  off, 32);
        sq += __shfl_xor(sq, off, 32);
    }
    float mu   = s * (1.0f / CH);
    float var  = sq * (1.0f / CH) - mu * mu;
    float rstd = rsqrtf(var + LN_EPS);
    #pragma unroll
    for (int i = 0; i < 4; ++i) {
        int c = lane + 32 * i;
        rp[c] = (v[i] - mu) * rstd * gamma[c] + beta[c];
    }
}

extern "C" void kernel_launch(void* const* d_in, const int* in_sizes, int n_in,
                              void* d_out, int out_size, void* d_ws, size_t ws_size,
                              hipStream_t stream) {
    const float* x        = (const float*)d_in[0];
    const int*   ei       = (const int*)  d_in[1];
    const float* lin_w    = (const float*)d_in[2];
    const float* att_src  = (const float*)d_in[3];
    const float* att_dst  = (const float*)d_in[4];
    const float* gat_bias = (const float*)d_in[5];
    const float* proj_w   = (const float*)d_in[6];
    const float* proj_b   = (const float*)d_in[7];
    const float* ln_g     = (const float*)d_in[8];
    const float* ln_b     = (const float*)d_in[9];

    int N = in_sizes[0] / CH;
    int E = in_sizes[1] / 2;
    int EN = E + N;

    // workspace layout (floats): h | a_src | a_dst | denom | maxkey | agg
    float*    h      = (float*)d_ws;                       // N*128
    float*    a_src  = h     + (size_t)N * CH;             // N*8
    float*    a_dst  = a_src + (size_t)N * HEADS;          // N*8
    float*    denom  = a_dst + (size_t)N * HEADS;          // N*8
    unsigned* maxkey = (unsigned*)(denom + (size_t)N * HEADS); // N*8
    float*    agg    = (float*)(maxkey + (size_t)N * HEADS);   // N*128
    float*    out    = (float*)d_out;

    int tiles      = (N + 15) / 16;
    int gemmBlocks = (tiles + 7) / 8;

    gat_gemm128_wmma<false><<<gemmBlocks, 256, 0, stream>>>(x, lin_w, nullptr,
                                                            nullptr, h, N);
    gat_att_coef<<<(N * HEADS + 255) / 256, 256, 0, stream>>>(h, att_src, att_dst,
                                                              a_src, a_dst, N);
    gat_init<<<(N * CH + 255) / 256, 256, 0, stream>>>(agg, denom, maxkey, N);
    gat_edge_max<<<(EN * HEADS + 255) / 256, 256, 0, stream>>>(ei, a_src, a_dst,
                                                               maxkey, E, N);
    gat_edge_denom<<<(EN * HEADS + 255) / 256, 256, 0, stream>>>(ei, a_src, a_dst,
                                                                 maxkey, denom, E, N);
    gat_edge_aggregate<<<(EN + 7) / 8, 256, 0, stream>>>(ei, h, a_src, a_dst,
                                                         maxkey, denom, agg, E, N);
    gat_gemm128_wmma<true><<<gemmBlocks, 256, 0, stream>>>(agg, proj_w, gat_bias,
                                                           proj_b, out, N);
    gat_layernorm<<<(N + 7) / 8, 256, 0, stream>>>(out, ln_g, ln_b, N);
}
